// SS2D_55490977464367
// MI455X (gfx1250) — matrix operand order repaired and checked by reading, not compiled, not run on hardware
//
#include <hip/hip_runtime.h>
#include <hip/hip_bf16.h>

// ---------------- problem constants ----------------
constexpr int BB   = 2;
constexpr int HH   = 64;
constexpr int WW   = 64;
constexpr int LL   = HH * WW;          // 4096
constexpr int DM   = 96;               // d_model
constexpr int DI   = 192;              // d_inner
constexpr int NS   = 16;               // d_state
constexpr int DTR  = 6;                // dt_rank
constexpr int NXP  = DTR + 2 * NS;     // 38 rows of x_proj_w
constexpr int CBC  = 2 * NS;           // 32 (B then C)

typedef __attribute__((ext_vector_type(16))) _Float16 v16h;
typedef __attribute__((ext_vector_type(8)))  float    v8f;

// ---------------- WMMA fragment helper ----------------
// Build a v16h fragment: elems 0..7 <- 8 f32 at p0, elems 8..15 <- 8 f32 at p1.
__device__ __forceinline__ v16h frag8x2(const float* p0, const float* p1) {
    float4 a0 = *reinterpret_cast<const float4*>(p0);
    float4 a1 = *reinterpret_cast<const float4*>(p0 + 4);
    float4 b0 = *reinterpret_cast<const float4*>(p1);
    float4 b1 = *reinterpret_cast<const float4*>(p1 + 4);
    v16h v;
    v[0]  = (_Float16)a0.x; v[1]  = (_Float16)a0.y; v[2]  = (_Float16)a0.z; v[3]  = (_Float16)a0.w;
    v[4]  = (_Float16)a1.x; v[5]  = (_Float16)a1.y; v[6]  = (_Float16)a1.z; v[7]  = (_Float16)a1.w;
    v[8]  = (_Float16)b0.x; v[9]  = (_Float16)b0.y; v[10] = (_Float16)b0.z; v[11] = (_Float16)b0.w;
    v[12] = (_Float16)b1.x; v[13] = (_Float16)b1.y; v[14] = (_Float16)b1.z; v[15] = (_Float16)b1.w;
    return v;
}

// ---------------- generic WMMA f16 GEMM ----------------
// C[m][n] = sum_k A0[m*srowA + k] * Bm[n*ldb + k]    (srowA may be negative: flipped scan dir)
// One wave computes a 16 x (NT*16) strip: A fragment loaded once per K-step, fed to NT WMMAs.
// K is compile-time -> full unroll, back-to-back v_wmma issue.
template<int K, int NT>
__global__ __launch_bounds__(256) void gemm_wmma(
    const float* __restrict__ A0, int srowA,
    const float* __restrict__ Bm, int ldb,
    float* __restrict__ C, int ldc,
    int Mtiles, int NSt)
{
    int tile = blockIdx.x * blockDim.y + threadIdx.y;
    if (tile >= Mtiles * NSt) return;              // wave-uniform: EXEC stays all-1s for WMMA
    int mt = tile / NSt, nst = tile % NSt;
    int lane = threadIdx.x;
    int hs = lane >> 4;        // half-select: 0 for lanes 0-15, 1 for 16-31
    int lm = lane & 15;
    const float* ar  = A0 + (long)(mt * 16 + lm) * (long)srowA;        // A row m
    const float* br0 = Bm + (long)(nst * NT * 16 + lm) * (long)ldb;    // B column block

    v8f acc[NT];
    #pragma unroll
    for (int j = 0; j < NT; ++j) acc[j] = (v8f){0.f,0.f,0.f,0.f,0.f,0.f,0.f,0.f};

    #pragma unroll
    for (int kb = 0; kb < K; kb += 32) {
        // A 16x32 f16 layout: lanes<16 -> K kb+0..7 / kb+16..23 ; lanes>=16 -> +8
        v16h af = frag8x2(ar + kb + hs * 8, ar + kb + 16 + hs * 8);
        #pragma unroll
        for (int j = 0; j < NT; ++j) {
            // B 32x16 f16 layout: lane holds column n, lanes<16 -> K kb+0..7 / kb+16..23 ; lanes>=16 -> +8
            const float* br = br0 + (long)j * 16 * ldb;
            v16h bf = frag8x2(br + kb + hs * 8, br + kb + 16 + hs * 8);
            acc[j] = __builtin_amdgcn_wmma_f32_16x16x32_f16(
                /*neg_a=*/false, af, /*neg_b=*/false, bf,
                /*c_mod=*/(short)0, acc[j], /*reuse_a=*/false, /*reuse_b=*/false);
        }
    }
    // C/D layout: acc elem r -> row mt*16 + hs*8 + r, col nst*NT*16 + j*16 + lm
    #pragma unroll
    for (int j = 0; j < NT; ++j) {
        float* crow = C + (long)(mt * 16 + hs * 8) * (long)ldc + (nst * NT * 16 + j * 16 + lm);
        #pragma unroll
        for (int r = 0; r < 8; ++r) crow[(long)r * ldc] = acc[j][r];
    }
}

// ---------------- depthwise 3x3 conv + bias + SiLU ----------------
// in: xz (b, l, 384) channels 0..191 ; out: xT (b, l, 192) row-major (transposed for GEMM/scan)
__global__ __launch_bounds__(256) void conv_silu_kernel(
    const float* __restrict__ xz, const float* __restrict__ cw,
    const float* __restrict__ cb, float* __restrict__ xT)
{
    int idx = blockIdx.x * 256 + threadIdx.x;
    if (idx >= BB * LL * DI) return;
    int d = idx % DI;
    int l = (idx / DI) % LL;
    int b = idx / (DI * LL);
    int h = l >> 6, w = l & 63;
    const float* base = xz + (long)b * LL * (2 * DI);
    float acc = cb[d];
    #pragma unroll
    for (int ky = 0; ky < 3; ++ky) {
        int hy = h + ky - 1;
        if (hy < 0 || hy >= HH) continue;
        #pragma unroll
        for (int kx = 0; kx < 3; ++kx) {
            int wx = w + kx - 1;
            if (wx < 0 || wx >= WW) continue;
            acc += base[(long)(hy * WW + wx) * (2 * DI) + d] * cw[d * 9 + ky * 3 + kx];
        }
    }
    float s = acc / (1.f + __expf(-acc));   // SiLU
    xT[(long)(b * LL + l) * DI + d] = s;
}

// ---------------- fold dt_proj into x_proj: Wc[k][d][dd] = sum_r dtw[k,d,r]*xpw[k,r,dd] ----------------
__global__ __launch_bounds__(256) void wcomb_kernel(
    const float* __restrict__ xpw, const float* __restrict__ dtw, float* __restrict__ Wc)
{
    int idx = blockIdx.x * 256 + threadIdx.x;
    if (idx >= 2 * DI * DI) return;
    int dd = idx % DI;
    int d  = (idx / DI) % DI;
    int k  = idx / (DI * DI);
    float acc = 0.f;
    #pragma unroll
    for (int r = 0; r < DTR; ++r)
        acc += dtw[(long)(k * DI + d) * DTR + r] * xpw[(long)(k * NXP + r) * DI + dd];
    Wc[idx] = acc;   // idx == (k*DI + d)*DI + dd
}

// ---------------- selective scan ----------------
// 24 blocks x 32 lanes: block = (b, dir k, channel-group g); lane owns one channel d,
// 16-state recurrence lives in registers; B_t/C_t broadcast via LDS.
__global__ __launch_bounds__(32) void scan_kernel(
    const float* __restrict__ dts, const float* __restrict__ xT,
    const float* __restrict__ BCm, const float* __restrict__ A_logs,
    const float* __restrict__ dtb, const float* __restrict__ Dsv,
    float* __restrict__ ys)
{
    int lane = threadIdx.x;
    int blk = blockIdx.x;
    int g = blk % 6;
    int k = (blk / 6) & 1;
    int b = blk / 12;
    int d = g * 32 + lane;
    int row = k * DI + d;

    float Ar[NS];
    #pragma unroll
    for (int n = 0; n < NS; ++n) Ar[n] = -__expf(A_logs[(long)row * NS + n]);
    float bias = dtb[row];
    float Dk   = Dsv[row];
    float h[NS];
    #pragma unroll
    for (int n = 0; n < NS; ++n) h[n] = 0.f;

    const float* dts_b = dts + (long)(b * 2 + k) * LL * DI;
    const float* bc_b  = BCm + (long)(b * 2 + k) * LL * CBC;
    float*       ys_b  = ys  + (long)(b * 2 + k) * LL * DI;
    const float* xT_b  = xT  + (long)b * LL * DI;

    __shared__ float sBC[CBC];
    for (int t = 0; t < LL; ++t) {
        float dtv = dts_b[(long)t * DI + d] + bias;
        dtv = (dtv > 20.f) ? dtv : log1pf(__expf(dtv));          // softplus
        int le = k ? (LL - 1 - t) : t;
        float u = xT_b[(long)le * DI + d];
        sBC[lane] = bc_b[(long)t * CBC + lane];
        __syncthreads();
        float du = dtv * u;
        float y = Dk * u;
        #pragma unroll
        for (int n = 0; n < NS; ++n) {
            float dA = __expf(dtv * Ar[n]);
            h[n] = dA * h[n] + du * sBC[n];
            y += h[n] * sBC[NS + n];
        }
        ys_b[(long)t * DI + d] = y;
        __syncthreads();
    }
}

// ---------------- combine directions + LayerNorm + SiLU(z) gate ----------------
__device__ __forceinline__ float block_sum_192(float v, float* s, int d) {
    s[d] = v; __syncthreads();
    #pragma unroll
    for (int n = 96; n >= 3; n >>= 1) {
        if (d < n) s[d] += s[d + n];
        __syncthreads();
    }
    if (d == 0) s[0] = s[0] + s[1] + s[2];
    __syncthreads();
    float r = s[0];
    __syncthreads();
    return r;
}

__global__ __launch_bounds__(192) void ln_gate_kernel(
    const float* __restrict__ ys, const float* __restrict__ xz,
    const float* __restrict__ lns, const float* __restrict__ lnb,
    float* __restrict__ yg)
{
    __shared__ float s[DI];
    int bl = blockIdx.x;           // 0 .. B*L-1
    int b = bl / LL, l = bl % LL;
    int d = threadIdx.x;
    float y = ys[((long)(b * 2 + 0) * LL + l) * DI + d]
            + ys[((long)(b * 2 + 1) * LL + (LL - 1 - l)) * DI + d];
    float mu = block_sum_192(y, s, d) * (1.f / DI);
    float dv = y - mu;
    float var = block_sum_192(dv * dv, s, d) * (1.f / DI);
    float yn = dv * rsqrtf(var + 1e-5f) * lns[d] + lnb[d];
    float z = xz[((long)(b * LL + l)) * (2 * DI) + DI + d];
    float gate = z / (1.f + __expf(-z));
    yg[((long)(b * LL + l)) * DI + d] = yn * gate;
}

// ---------------- host orchestration ----------------
extern "C" void kernel_launch(void* const* d_in, const int* in_sizes, int n_in,
                              void* d_out, int out_size, void* d_ws, size_t ws_size,
                              hipStream_t stream) {
    const float* x    = (const float*)d_in[0];   // (B,H,W,96)
    const float* ipw  = (const float*)d_in[1];   // (384,96)
    const float* cw   = (const float*)d_in[2];   // (192,1,3,3)
    const float* cb   = (const float*)d_in[3];   // (192,)
    const float* xpw  = (const float*)d_in[4];   // (2,38,192)
    const float* dtw  = (const float*)d_in[5];   // (2,192,6)
    const float* dtb  = (const float*)d_in[6];   // (2,192)
    const float* alog = (const float*)d_in[7];   // (384,16)
    const float* Dsv  = (const float*)d_in[8];   // (384,)
    const float* lns  = (const float*)d_in[9];   // (192,)
    const float* lnb  = (const float*)d_in[10];  // (192,)
    const float* opw  = (const float*)d_in[11];  // (96,192)
    float* out = (float*)d_out;                  // (B,H,W,96)

    float* ws  = (float*)d_ws;
    float* xz  = ws;                             // (B*L, 384)   3,145,728 f
    float* xT  = xz  + (long)BB * LL * 2 * DI;   // (B*L, 192)   1,572,864 f
    float* Wc  = xT  + (long)BB * LL * DI;       // (2,192,192)     73,728 f
    float* BCb = Wc  + 2 * DI * DI;              // (B*2, L, 32) 524,288 f
    float* dts = BCb + (long)BB * 2 * LL * CBC;  // (B*2, L,192) 3,145,728 f
    float* ysb = dts + (long)BB * 2 * LL * DI;   // (B*2, L,192) 3,145,728 f
    float* yg  = dts;                            // alias: dts fully consumed by scan

    dim3 gblk(32, 8);
    const int ML = BB * LL / 16;                 // 512 M-tiles for full-batch GEMMs

    // 1) in_proj: (8192 x 384) = x (8192x96) * in_proj_w^T ; N = 6 supertiles of 64
    gemm_wmma<DM, 4><<<dim3((ML * 6) / 8), gblk, 0, stream>>>(
        x, DM, ipw, DM, xz, 2 * DI, ML, 6);

    // 2) depthwise conv + SiLU -> xT (b,l,d)
    conv_silu_kernel<<<dim3((BB * LL * DI) / 256), dim3(256), 0, stream>>>(xz, cw, cb, xT);

    // 3) fold dt_proj into x_proj
    wcomb_kernel<<<dim3((2 * DI * DI + 255) / 256), dim3(256), 0, stream>>>(xpw, dtw, Wc);

    // 4) per (b,dir): B/C projection (Lx32) and fused dt projection (Lx192); dir=1 uses
    //    a negative A row stride to realize the sequence flip with zero extra copies.
    for (int b = 0; b < BB; ++b) {
        for (int k = 0; k < 2; ++k) {
            const float* A0 = xT + (long)b * LL * DI + (k ? (long)(LL - 1) * DI : 0);
            int srow = k ? -DI : DI;
            gemm_wmma<DI, 2><<<dim3((LL / 16) / 8), gblk, 0, stream>>>(         // N=32: 1 supertile
                A0, srow, xpw + (long)(k * NXP + DTR) * DI, DI,
                BCb + (long)(b * 2 + k) * LL * CBC, CBC, LL / 16, 1);
            gemm_wmma<DI, 4><<<dim3((LL / 16 * 3) / 8), gblk, 0, stream>>>(     // N=192: 3 supertiles
                A0, srow, Wc + (long)k * DI * DI, DI,
                dts + (long)(b * 2 + k) * LL * DI, DI, LL / 16, 3);
        }
    }

    // 5) selective scan: 24 blocks (b,k,channel-group) x wave32
    scan_kernel<<<dim3(BB * 2 * (DI / 32)), dim3(32), 0, stream>>>(
        dts, xT, BCb, alog, dtb, Dsv, ysb);

    // 6) combine dirs + LayerNorm + gate
    ln_gate_kernel<<<dim3(BB * LL), dim3(DI), 0, stream>>>(ysb, xz, lns, lnb, yg);

    // 7) out_proj: (8192 x 96) = yg (8192x192) * out_proj_w^T -> d_out ; N = 3 supertiles of 32
    gemm_wmma<DI, 2><<<dim3((ML * 3) / 8), gblk, 0, stream>>>(
        yg, DI, opw, DI, out, DM, ML, 3);

    (void)in_sizes; (void)n_in; (void)out_size; (void)ws_size;
}
